// EnhancedSelfAttention_4389456576577
// MI455X (gfx1250) — matrix operand order repaired and checked by reading, compiled, not run on hardware
//
#include <hip/hip_runtime.h>
#include <hip/hip_bf16.h>
#include <math.h>

// ---------------------------------------------------------------------------
// EnhancedSelfAttention for MI455X (gfx1250), wave32 + WMMA bf16 + async-LDS.
//
//   cvt:   hidden, Wq, Wk, Wv, Wo -> bf16 (one-time)
//   bias:  per-head T5 relative-position bias table tbl[h][q-k+1023]
//          (kills the v_log_f32 chain in the attention hot loop)
//   gemm:  bf16 WMMA 16x16x32, double-buffered global_load_async_to_lds_b128
//          staging (s_wait_asynccnt 4 overlaps copy of tile i+1 with compute)
//   attn:  per (b,h,64 q-rows): QK^T WMMA + bias + mask, online softmax
//          stats, recompute pass -> attn_weights (f32) + P*V WMMA -> bf16
// ---------------------------------------------------------------------------

typedef __attribute__((ext_vector_type(16))) __bf16 v16bf;
typedef __attribute__((ext_vector_type(8)))  float  v8f;

union FragBF {
    v16bf v;
    unsigned short u[16];
    uint4 q[2];
};

__device__ __forceinline__ unsigned short f2bf(float f) {
    unsigned int u = __float_as_uint(f);
    unsigned int r = u + 0x7FFFu + ((u >> 16) & 1u);   // round-to-nearest-even
    return (unsigned short)(r >> 16);
}

// CDNA5 async copy: 16B global -> LDS, no VGPR round trip, tracked by ASYNCcnt.
#define ASYNC_CP16(ldsoff, gptr)                                              \
    asm volatile("global_load_async_to_lds_b128 %0, %1, off"                  \
                 :: "v"(ldsoff), "v"(gptr) : "memory")
#define ASYNC_WAIT0() asm volatile("s_wait_asynccnt 0" ::: "memory")
#define ASYNC_WAIT4() asm volatile("s_wait_asynccnt 4" ::: "memory")

__device__ __forceinline__ unsigned lds_off(const void* p) {
    return (unsigned)(unsigned long long)p;
}

// A-matrix fragment, 16x32 bf16 (ISA 7.12.2). Needs 16B-aligned rows.
__device__ __forceinline__ void load_frag_a16(const unsigned short* base, int stride,
                                              int lane, FragBF& f) {
    const unsigned short* p = base + (lane & 15) * stride + ((lane >> 4) * 8);
    f.q[0] = *(const uint4*)(p);
    f.q[1] = *(const uint4*)(p + 16);
}

// B-matrix fragment, 32x16 bf16, source stored [n][k] row-major (K contiguous).
__device__ __forceinline__ void load_frag_b16_rowk(const unsigned short* base, int stride,
                                                   int lane, FragBF& f) {
    const unsigned short* p = base + (lane & 15) * stride + (lane >> 4) * 8;
    f.q[0] = *(const uint4*)(p);
    f.q[1] = *(const uint4*)(p + 16);
}

__device__ __forceinline__ float wave_max32(float v) {
#pragma unroll
    for (int o = 16; o > 0; o >>= 1) v = fmaxf(v, __shfl_xor(v, o, 32));
    return v;
}
__device__ __forceinline__ float wave_sum32(float v) {
#pragma unroll
    for (int o = 16; o > 0; o >>= 1) v += __shfl_xor(v, o, 32);
    return v;
}

// ---------------------------------------------------------------------------
// fp32 -> bf16 pre-conversion (one-time per tensor)
// ---------------------------------------------------------------------------
__global__ __launch_bounds__(256) void cvt_f32_bf16_kernel(
    const float* __restrict__ s, unsigned short* __restrict__ d, int n)
{
    const int i = (blockIdx.x * 256 + threadIdx.x) * 4;
    if (i + 3 < n) {
        const float4 v = *(const float4*)(s + i);
        d[i + 0] = f2bf(v.x);
        d[i + 1] = f2bf(v.y);
        d[i + 2] = f2bf(v.z);
        d[i + 3] = f2bf(v.w);
    }
}

// ---------------------------------------------------------------------------
// T5 relative-position bias table: tbl[h][d + 1023], d = qpos - kpos
// ---------------------------------------------------------------------------
__device__ __forceinline__ int rel_bucket_from_delta(int n /* = qpos - kpos */) {
    const int ret = (n < 0) ? 16 : 0;
    n = (n < 0) ? -n : n;
    int b;
    if (n < 8) {
        b = n;
    } else {
        const float t = __logf((float)n * 0.125f) * (8.0f / 2.772588722f);
        const int vl = 8 + (int)t;
        b = (vl < 15) ? vl : 15;
    }
    return ret + b;
}

__global__ __launch_bounds__(256) void bias_table_kernel(
    const float* __restrict__ relb, float* __restrict__ tbl)
{
    const int h = blockIdx.x;                  // 16 heads
    for (int i = threadIdx.x; i < 2048; i += 256) {
        const int d = (i < 2047) ? (i - 1023) : 0;   // pad entry harmless
        const int bkt = rel_bucket_from_delta(d);
        tbl[h * 2048 + i] = relb[bkt * 16 + h];
    }
}

// ---------------------------------------------------------------------------
// GEMM: Out[M,N] = X[M,K] * W[N,K]^T, bf16 operands, WMMA f32 accumulate.
// Block tile 128x128, K-step 32, 256 threads = 8 waves, double-buffered LDS.
// ---------------------------------------------------------------------------
template <bool OUT_BF16>
__global__ __launch_bounds__(256) void gemm_bf16_kernel(
    const unsigned short* __restrict__ X, const unsigned short* __restrict__ W,
    void* __restrict__ OutP, int M, int N, int K)
{
    __shared__ __align__(16) unsigned short As[2][128][40];  // 80B rows
    __shared__ __align__(16) unsigned short Bs[2][128][40];

    const int tid  = threadIdx.x;
    const int lane = tid & 31;
    const int wv   = tid >> 5;
    const int m0   = blockIdx.y * 128;
    const int n0   = blockIdx.x * 128;

    const v8f vzero = {0.f, 0.f, 0.f, 0.f, 0.f, 0.f, 0.f, 0.f};
    v8f acc[8];
#pragma unroll
    for (int j = 0; j < 8; ++j) acc[j] = vzero;

    auto issue_tile = [&](int buf, int k0) {   // 4 async 16B copies per thread
#pragma unroll
        for (int cc = 0; cc < 2; ++cc) {
            const int c    = tid + cc * 256;
            const int row  = c >> 2;
            const int part = c & 3;
            ASYNC_CP16(lds_off(&As[buf][row][part * 8]),
                       X + (size_t)(m0 + row) * K + k0 + part * 8);
            ASYNC_CP16(lds_off(&Bs[buf][row][part * 8]),
                       W + (size_t)(n0 + row) * K + k0 + part * 8);
        }
    };

    const int nIter = K / 32;
    issue_tile(0, 0);
    for (int it = 0; it < nIter; ++it) {
        const int buf = it & 1;
        if (it + 1 < nIter) {
            issue_tile(buf ^ 1, (it + 1) * 32);
            ASYNC_WAIT4();                      // oldest 4 = this tile's copies
        } else {
            ASYNC_WAIT0();
        }
        __syncthreads();

        FragBF a;
        load_frag_a16(&As[buf][wv * 16][0], 40, lane, a);
#pragma unroll
        for (int j = 0; j < 8; ++j) {
            FragBF b;
            load_frag_b16_rowk(&Bs[buf][j * 16][0], 40, lane, b);
            acc[j] = __builtin_amdgcn_wmma_f32_16x16x32_bf16(
                false, a.v, false, b.v, (short)0, acc[j], false, false);
        }
        __syncthreads();                        // before next issue reuses buf
    }

    const int mrow  = m0 + wv * 16 + 8 * (lane >> 4);
    const int ncol0 = n0 + (lane & 15);
    if (OUT_BF16) {
        unsigned short* Out = (unsigned short*)OutP;
#pragma unroll
        for (int j = 0; j < 8; ++j)
#pragma unroll
            for (int e = 0; e < 8; ++e)
                Out[(size_t)(mrow + e) * N + ncol0 + j * 16] = f2bf(acc[j][e]);
    } else {
        float* Out = (float*)OutP;
#pragma unroll
        for (int j = 0; j < 8; ++j)
#pragma unroll
            for (int e = 0; e < 8; ++e)
                Out[(size_t)(mrow + e) * N + ncol0 + j * 16] = acc[j][e];
    }
}

// ---------------------------------------------------------------------------
// Attention
// ---------------------------------------------------------------------------
#define SEQ   1024
#define HEADS 16
#define DKV   64
#define TQ    64
#define NCH   16

__global__ __launch_bounds__(256) void attn_kernel(
    const unsigned short* __restrict__ Q, const unsigned short* __restrict__ K,
    const unsigned short* __restrict__ V, const int* __restrict__ mask,
    const float* __restrict__ btab,
    float* __restrict__ attnW, unsigned short* __restrict__ attnO)
{
    __shared__ __align__(16) unsigned short Qs[TQ][72];  // 144B rows
    __shared__ __align__(16) unsigned short Ks[64][72];
    __shared__ __align__(16) unsigned short Vt[64][72];  // transposed [dk][kc]
    __shared__ __align__(16) unsigned short Ps[TQ][72];
    __shared__ __align__(16) float          Sb[TQ][65];
    __shared__ __align__(16) float          biasL[2048]; // tbl row for head h
    __shared__ float rowM[TQ];
    __shared__ float rowL[TQ];

    const int tid  = threadIdx.x;
    const int lane = tid & 31;
    const int wv   = tid >> 5;

    const int bid = blockIdx.x;                // B*H*(S/TQ) = 1024
    const int b   = bid >> 8;
    const int h   = (bid >> 4) & 15;
    const int q0  = (bid & 15) * TQ;

    const int lr  = tid >> 2;                  // staging row
    const int lc0 = (tid & 3) * 16;            // staging col base

    // stage Q (512 x 16B) and bias row (512 x 16B) asynchronously
#pragma unroll
    for (int cc = 0; cc < 2; ++cc) {
        const int c    = tid + cc * 256;
        const int row  = c >> 3;
        const int part = c & 7;
        ASYNC_CP16(lds_off(&Qs[row][part * 8]),
                   Q + (size_t)(b * SEQ + q0 + row) * 1024 + h * DKV + part * 8);
        ASYNC_CP16(lds_off(&biasL[c * 4]), btab + h * 2048 + c * 4);
    }

    const v8f vzero = {0.f, 0.f, 0.f, 0.f, 0.f, 0.f, 0.f, 0.f};

    float m_i[8], l_i[8];                      // wave wv owns rows [wv*8, +8)
#pragma unroll
    for (int r = 0; r < 8; ++r) { m_i[r] = -3.0e38f; l_i[r] = 0.f; }

    auto issue_k = [&](int c0) {
#pragma unroll
        for (int cc = 0; cc < 2; ++cc) {
            const int c    = tid + cc * 256;
            const int row  = c >> 3;
            const int part = c & 7;
            ASYNC_CP16(lds_off(&Ks[row][part * 8]),
                       K + (size_t)(b * SEQ + c0 + row) * 1024 + h * DKV + part * 8);
        }
    };

    // score tiles for one 64-col K-chunk -> Sb (scale + bias + mask applied)
    auto score_pass = [&](int c0) {
#pragma unroll
        for (int tt = 0; tt < 2; ++tt) {
            const int t  = wv * 2 + tt;
            const int qi = t >> 2;
            const int ki = t & 3;
            v8f acc = vzero;
#pragma unroll
            for (int s = 0; s < 2; ++s) {
                FragBF a, bfr;
                load_frag_a16(&Qs[qi * 16][s * 32], 72, lane, a);
                load_frag_b16_rowk(&Ks[ki * 16][s * 32], 72, lane, bfr);
                acc = __builtin_amdgcn_wmma_f32_16x16x32_bf16(
                    false, a.v, false, bfr.v, (short)0, acc, false, false);
            }
            const int qrl  = qi * 16 + 8 * (lane >> 4);
            const int kcl  = ki * 16 + (lane & 15);
            const int kabs = c0 + kcl;
            const float mterm = (mask[b * SEQ + kabs] == 0) ? -1e9f : 0.f;
            const int ib0 = q0 + qrl - kabs + 1023;   // bias idx for e=0
#pragma unroll
            for (int e = 0; e < 8; ++e)
                Sb[qrl + e][kcl] = acc[e] * 0.125f + biasL[ib0 + e] + mterm;
        }
    };

    // -------- pass 1: per-row max / sum-of-exp (online) --------
    issue_k(0);
    for (int c = 0; c < NCH; ++c) {
        ASYNC_WAIT0();
        __syncthreads();                       // Ks (and first time: Q/bias) ready
        score_pass(c * 64);
        __syncthreads();                       // Sb ready, Ks free
        if (c + 1 < NCH) issue_k((c + 1) * 64);// overlap copy with stats
#pragma unroll
        for (int rr = 0; rr < 8; ++rr) {
            const int row = wv * 8 + rr;
            const float x0 = Sb[row][lane];
            const float x1 = Sb[row][lane + 32];
            const float tmax = wave_max32(fmaxf(x0, x1));
            const float mn   = fmaxf(m_i[rr], tmax);
            const float ss   = wave_sum32(__expf(x0 - mn) + __expf(x1 - mn));
            l_i[rr] = l_i[rr] * __expf(m_i[rr] - mn) + ss;
            m_i[rr] = mn;
        }
        __syncthreads();
    }
    if (lane == 0) {
#pragma unroll
        for (int rr = 0; rr < 8; ++rr) {
            rowM[wv * 8 + rr] = m_i[rr];
            rowL[wv * 8 + rr] = l_i[rr];
        }
    }
    __syncthreads();

    // -------- pass 2: recompute scores, emit weights, accumulate P*V --------
    v8f oacc[2] = {vzero, vzero};
    issue_k(0);
    for (int c = 0; c < NCH; ++c) {
        const int c0 = c * 64;
        ASYNC_WAIT0();
        __syncthreads();                       // Ks ready
        score_pass(c0);
        {                                      // stage V chunk transposed
            const unsigned short* vp =
                V + (size_t)(b * SEQ + c0 + lr) * 1024 + h * DKV + lc0;
#pragma unroll
            for (int i = 0; i < 16; ++i) Vt[lc0 + i][lr] = vp[i];
        }
        __syncthreads();                       // Sb + Vt ready, Ks free
        {
            const float mr   = rowM[lr];
            const float linv = 1.0f / rowL[lr];
            float* wout = attnW + ((size_t)((b * HEADS + h) * SEQ + q0 + lr)) * SEQ
                                 + c0 + lc0;
#pragma unroll
            for (int i = 0; i < 16; ++i) {
                const float p = __expf(Sb[lr][lc0 + i] - mr) * linv;
                wout[i] = p;
                Ps[lr][lc0 + i] = f2bf(p);
            }
        }
        if (c + 1 < NCH) issue_k((c + 1) * 64);// overlap copy with PV
        __syncthreads();                       // Ps ready
#pragma unroll
        for (int tt = 0; tt < 2; ++tt) {
            const int t  = wv * 2 + tt;
            const int qi = t >> 2;
            const int di = t & 3;
#pragma unroll
            for (int s = 0; s < 2; ++s) {
                FragBF a, bb;
                load_frag_a16(&Ps[qi * 16][s * 32], 72, lane, a);
                load_frag_b16_rowk(&Vt[di * 16][s * 32], 72, lane, bb);
                oacc[tt] = __builtin_amdgcn_wmma_f32_16x16x32_bf16(
                    false, a.v, false, bb.v, (short)0, oacc[tt], false, false);
            }
        }
        __syncthreads();                       // before Vt/Ps reuse
    }

    // write attention output [B,S,H*DKV] as bf16 (feeds final projection GEMM)
#pragma unroll
    for (int tt = 0; tt < 2; ++tt) {
        const int t   = wv * 2 + tt;
        const int qi  = t >> 2;
        const int di  = t & 3;
        const int qrl = qi * 16 + 8 * (lane >> 4);
        const int dc  = di * 16 + (lane & 15);
#pragma unroll
        for (int e = 0; e < 8; ++e)
            attnO[(size_t)(b * SEQ + q0 + qrl + e) * 1024 + h * DKV + dc] =
                f2bf(oacc[tt][e]);
    }
}

// ---------------------------------------------------------------------------
extern "C" void kernel_launch(void* const* d_in, const int* in_sizes, int n_in,
                              void* d_out, int out_size, void* d_ws, size_t ws_size,
                              hipStream_t stream) {
    (void)in_sizes; (void)n_in; (void)out_size; (void)ws_size;

    const float* hidden = (const float*)d_in[0];   // [4,1024,1024]
    const int*   mask   = (const int*)  d_in[1];   // [4,1,1,1024]
    const float* Wq     = (const float*)d_in[2];   // [1024,1024]
    const float* Wk     = (const float*)d_in[3];
    const float* Wv     = (const float*)d_in[4];
    const float* Wo     = (const float*)d_in[5];
    const float* relb   = (const float*)d_in[6];   // [32,16]

    const size_t BSD = (size_t)4 * 1024 * 1024;
    const size_t WSZ = (size_t)1024 * 1024;

    float* out   = (float*)d_out;                  // [4,1024,1024]
    float* attnW = out + BSD;                      // [4,16,1024,1024]

    unsigned short* hb  = (unsigned short*)d_ws;   // bf16 scratch
    unsigned short* wqb = hb  + BSD;
    unsigned short* wkb = wqb + WSZ;
    unsigned short* wvb = wkb + WSZ;
    unsigned short* wob = wvb + WSZ;
    unsigned short* qb  = wob + WSZ;
    unsigned short* kb  = qb  + BSD;
    unsigned short* vb  = kb  + BSD;
    unsigned short* ob  = vb  + BSD;
    float* btab = (float*)(ob + BSD);              // [16][2048] bias table

    dim3 blk(256);

    cvt_f32_bf16_kernel<<<dim3((int)(BSD / 1024)), blk, 0, stream>>>(hidden, hb, (int)BSD);
    cvt_f32_bf16_kernel<<<dim3((int)(WSZ / 1024)), blk, 0, stream>>>(Wq, wqb, (int)WSZ);
    cvt_f32_bf16_kernel<<<dim3((int)(WSZ / 1024)), blk, 0, stream>>>(Wk, wkb, (int)WSZ);
    cvt_f32_bf16_kernel<<<dim3((int)(WSZ / 1024)), blk, 0, stream>>>(Wv, wvb, (int)WSZ);
    cvt_f32_bf16_kernel<<<dim3((int)(WSZ / 1024)), blk, 0, stream>>>(Wo, wob, (int)WSZ);
    bias_table_kernel<<<dim3(16), blk, 0, stream>>>(relb, btab);

    dim3 gG(1024 / 128, 4096 / 128);               // (N/128, M/128)
    gemm_bf16_kernel<true><<<gG, blk, 0, stream>>>(hb, wqb, qb, 4096, 1024, 1024);
    gemm_bf16_kernel<true><<<gG, blk, 0, stream>>>(hb, wkb, kb, 4096, 1024, 1024);
    gemm_bf16_kernel<true><<<gG, blk, 0, stream>>>(hb, wvb, vb, 4096, 1024, 1024);

    attn_kernel<<<dim3(1024), blk, 0, stream>>>(qb, kb, vb, mask, btab, attnW, ob);

    gemm_bf16_kernel<false><<<gG, blk, 0, stream>>>(ob, wob, out, 4096, 1024, 1024);
}
